// TrackerTorch_75007308857870
// MI455X (gfx1250) — matrix operand order repaired and checked by reading, not compiled, not run on hardware
//
#include <hip/hip_runtime.h>
#include <hip/hip_bf16.h>

typedef __bf16 bf16_t;
typedef __attribute__((ext_vector_type(16))) __bf16 v16bf;
typedef __attribute__((ext_vector_type(8)))  __bf16 v8bf;
typedef __attribute__((ext_vector_type(8)))  float  v8f;

#define Z   512        // feature dim
#define ZT  (Z / 16)   // 32 column tiles over feature dim
#define CH  25         // anchor chunks for argmin partials
#define PSLOTS (CH * 4)

// ---------- WMMA fragment loaders (NT GEMM, row-major [rows, K] storage) ----------
// A 16x32 bf16 fragment (M x K): lane r=lane%16 is row M; h=lane/16 selects K phase.
// Per ISA 7.12.2: lane holds K = {kk+8h .. +7} and {kk+16+8h .. +7} (two 16B runs).
__device__ __forceinline__ v16bf load_a_frag(const bf16_t* __restrict__ row, int kk, int h) {
  v8bf lo = *(const v8bf*)(row + kk + 8 * h);
  v8bf hi = *(const v8bf*)(row + kk + 16 + 8 * h);
  v16bf a;
#pragma unroll
  for (int t = 0; t < 8; ++t) { a[t] = lo[t]; a[t + 8] = hi[t]; }
  return a;
}
// B 32x16 bf16 fragment (K x N): lane n=lane%16 is column N; lane holds
// K = {kk+8h .. +7} and {kk+16+8h .. +7} (two 16B runs), matching A.
__device__ __forceinline__ v16bf load_b_frag(const bf16_t* __restrict__ row, int kk, int h) {
  v8bf lo = *(const v8bf*)(row + kk + 8 * h);
  v8bf hi = *(const v8bf*)(row + kk + 16 + 8 * h);
  v16bf b;
#pragma unroll
  for (int t = 0; t < 8; ++t) { b[t] = lo[t]; b[t + 8] = hi[t]; }
  return b;
}

__device__ __forceinline__ v8f wmma_bf16(v16bf a, v16bf b, v8f c) {
  return __builtin_amdgcn_wmma_f32_16x16x32_bf16(false, a, false, b, (short)0, c, false, false);
}

// ---------- K0: L2-normalize rows, emit bf16 ----------
__global__ void k_rownorm_bf16(const float* __restrict__ in, bf16_t* __restrict__ out, int z) {
  __shared__ float red[256];
  int row = blockIdx.x, tid = threadIdx.x;
  const float* r = in + (size_t)row * z;
  float s = 0.f;
  for (int c = tid; c < z; c += 256) { float v = r[c]; s += v * v; }
  red[tid] = s; __syncthreads();
  for (int off = 128; off > 0; off >>= 1) {
    if (tid < off) red[tid] += red[tid + off];
    __syncthreads();
  }
  float inv = rsqrtf(red[0]);
  bf16_t* o = out + (size_t)row * z;
  for (int c = tid; c < z; c += 256) o[c] = (bf16_t)(r[c] * inv);
}

// ---------- K1: transpose x_bf16 [B,Z] -> xT [Z,B] ----------
__global__ void k_transpose(const bf16_t* __restrict__ in, bf16_t* __restrict__ out,
                            int rows, int cols) {
  int idx = blockIdx.x * 256 + threadIdx.x;
  if (idx >= rows * cols) return;
  int c = idx / rows, r = idx % rows;               // r fastest -> coalesced writes
  out[(size_t)c * rows + r] = in[(size_t)r * cols + c];
}

// ---------- K2: S = x @ x^T, threshold -> bf16 mask[B,B] ----------
// 4M x 2N register macro-tile per wave (64x32 output): 6 frag loads feed 8 WMMAs.
__global__ void k_self_mask(const bf16_t* __restrict__ xb, bf16_t* __restrict__ mask,
                            int n, int bt) {
  int wave = (blockIdx.x * blockDim.x + threadIdx.x) >> 5;
  int lane = threadIdx.x & 31;
  int mtiles = bt >> 2, ntiles = bt >> 1;
  int im = wave / ntiles, jn = wave % ntiles;
  if (im >= mtiles) return;                          // wave-uniform
  int r = lane & 15, h = lane >> 4;
  const bf16_t* arow[4];
  const bf16_t* brow[2];
#pragma unroll
  for (int mi = 0; mi < 4; ++mi) arow[mi] = xb + (size_t)((im * 4 + mi) * 16 + r) * Z;
#pragma unroll
  for (int ni = 0; ni < 2; ++ni) brow[ni] = xb + (size_t)((jn * 2 + ni) * 16 + r) * Z;
  v8f acc[4][2] = {};
#pragma unroll
  for (int kk = 0; kk < Z; kk += 32) {
    v16bf a[4], b[2];
#pragma unroll
    for (int mi = 0; mi < 4; ++mi) a[mi] = load_a_frag(arow[mi], kk, h);
#pragma unroll
    for (int ni = 0; ni < 2; ++ni) b[ni] = load_b_frag(brow[ni], kk, h);
#pragma unroll
    for (int mi = 0; mi < 4; ++mi)
#pragma unroll
      for (int ni = 0; ni < 2; ++ni)
        acc[mi][ni] = wmma_bf16(a[mi], b[ni], acc[mi][ni]);
  }
#pragma unroll
  for (int mi = 0; mi < 4; ++mi)
#pragma unroll
    for (int ni = 0; ni < 2; ++ni)
#pragma unroll
      for (int v = 0; v < 8; ++v) {                  // cos_d < 0.6 <=> dot > 0.4
        float m = (acc[mi][ni][v] > 0.4f) ? 1.0f : 0.0f;   // {0,1} exact in bf16
        mask[(size_t)((im * 4 + mi) * 16 + v + 8 * h) * n + (jn * 2 + ni) * 16 + r] = (bf16_t)m;
      }
}

// ---------- K3: adjusted_raw = mask @ x  (NT with B = xT rows), 4M x 2N blocked ----------
__global__ void k_mask_x(const bf16_t* __restrict__ mask, const bf16_t* __restrict__ xT,
                         float* __restrict__ adj, int n, int bt) {
  int wave = (blockIdx.x * blockDim.x + threadIdx.x) >> 5;
  int lane = threadIdx.x & 31;
  int mtiles = bt >> 2, ntn = ZT >> 1;               // 64 x 16 macro grid
  int im = wave / ntn, jn = wave % ntn;
  if (im >= mtiles) return;                          // wave-uniform
  int r = lane & 15, h = lane >> 4;
  const bf16_t* arow[4];
  const bf16_t* brow[2];
#pragma unroll
  for (int mi = 0; mi < 4; ++mi) arow[mi] = mask + (size_t)((im * 4 + mi) * 16 + r) * n;
#pragma unroll
  for (int ni = 0; ni < 2; ++ni) brow[ni] = xT + (size_t)((jn * 2 + ni) * 16 + r) * n;
  v8f acc[4][2] = {};
  for (int kk = 0; kk < n; kk += 32) {               // runtime K=4096
    v16bf a[4], b[2];
#pragma unroll
    for (int mi = 0; mi < 4; ++mi) a[mi] = load_a_frag(arow[mi], kk, h);
#pragma unroll
    for (int ni = 0; ni < 2; ++ni) b[ni] = load_b_frag(brow[ni], kk, h);
#pragma unroll
    for (int mi = 0; mi < 4; ++mi)
#pragma unroll
      for (int ni = 0; ni < 2; ++ni)
        acc[mi][ni] = wmma_bf16(a[mi], b[ni], acc[mi][ni]);
  }
#pragma unroll
  for (int mi = 0; mi < 4; ++mi)
#pragma unroll
    for (int ni = 0; ni < 2; ++ni)
#pragma unroll
      for (int v = 0; v < 8; ++v)
        adj[(size_t)((im * 4 + mi) * 16 + v + 8 * h) * Z + (jn * 2 + ni) * 16 + r] = acc[mi][ni][v];
}

// ---------- K4: count = rowsum(mask); adjusted /= count; -> bf16 ----------
__global__ void k_div_cvt(const bf16_t* __restrict__ mask, const float* __restrict__ adj,
                          bf16_t* __restrict__ adjb, int n) {
  __shared__ float red[256];
  int row = blockIdx.x, tid = threadIdx.x;
  float s = 0.f;
  const bf16_t* mrow = mask + (size_t)row * n;
  for (int j = tid; j < n; j += 256) s += (float)mrow[j];
  red[tid] = s; __syncthreads();
  for (int off = 128; off > 0; off >>= 1) {
    if (tid < off) red[tid] += red[tid + off];
    __syncthreads();
  }
  float inv = 1.0f / red[0];                         // diagonal guarantees >= 1
  for (int col = tid; col < Z; col += 256)
    adjb[(size_t)row * Z + col] = (bf16_t)(adj[(size_t)row * Z + col] * inv);
}

// ---------- K5: D = adjusted @ anchors_n^T, fused |0.7 - s| argmin, 2M x 2N ----------
__global__ void k_anchor_argmin(const bf16_t* __restrict__ adjb, const bf16_t* __restrict__ anb,
                                unsigned long long* __restrict__ partial, int at_total, int tpc) {
  int chunk = blockIdx.x;                            // 0..CH-1
  int itm = blockIdx.y;                              // macro i-tile (2 subtiles = 32 rows)
  int w = threadIdx.x >> 5;                          // wave 0..3
  int lane = threadIdx.x & 31;
  int r = lane & 15, h = lane >> 4;
  const bf16_t* arow[2];
#pragma unroll
  for (int mi = 0; mi < 2; ++mi) arow[mi] = adjb + (size_t)((itm * 2 + mi) * 16 + r) * Z;
  unsigned long long best[2][8];
#pragma unroll
  for (int mi = 0; mi < 2; ++mi)
#pragma unroll
    for (int v = 0; v < 8; ++v) best[mi][v] = 0xFFFFFFFFFFFFFFFFull;

  for (int t = 2 * w; t < tpc; t += 8) {             // wave-uniform; pairs of anchor tiles
    int at0 = chunk * tpc + t;
    if (at0 >= at_total) break;
    int at1 = (at0 + 1 < at_total && t + 1 < tpc) ? at0 + 1 : at0;  // clamp dup: idempotent under min
    int at[2] = { at0, at1 };
    const bf16_t* brow[2];
#pragma unroll
    for (int ni = 0; ni < 2; ++ni) brow[ni] = anb + (size_t)(at[ni] * 16 + r) * Z;
    v8f acc[2][2] = {};
#pragma unroll
    for (int kk = 0; kk < Z; kk += 32) {
      v16bf a[2], b[2];
#pragma unroll
      for (int mi = 0; mi < 2; ++mi) a[mi] = load_a_frag(arow[mi], kk, h);
#pragma unroll
      for (int ni = 0; ni < 2; ++ni) b[ni] = load_b_frag(brow[ni], kk, h);
#pragma unroll
      for (int mi = 0; mi < 2; ++mi)
#pragma unroll
        for (int ni = 0; ni < 2; ++ni)
          acc[mi][ni] = wmma_bf16(a[mi], b[ni], acc[mi][ni]);
    }
#pragma unroll
    for (int ni = 0; ni < 2; ++ni) {
      unsigned aidx = (unsigned)(at[ni] * 16 + r);
#pragma unroll
      for (int mi = 0; mi < 2; ++mi)
#pragma unroll
        for (int v = 0; v < 8; ++v) {
          float dist = fabsf(0.7f - acc[mi][ni][v]);           // |(1-s) - 0.3|
          unsigned long long p = ((unsigned long long)__float_as_uint(dist) << 32) | aidx;
          best[mi][v] = p < best[mi][v] ? p : best[mi][v];     // tie -> lower idx
        }
    }
  }
  // reduce across the 16 N-lanes (xor 1..8 stays within each 16-lane half)
#pragma unroll
  for (int m = 1; m < 16; m <<= 1)
#pragma unroll
    for (int mi = 0; mi < 2; ++mi)
#pragma unroll
      for (int v = 0; v < 8; ++v) {
        unsigned lo = __shfl_xor((unsigned)best[mi][v], m, 32);
        unsigned hi = __shfl_xor((unsigned)(best[mi][v] >> 32), m, 32);
        unsigned long long o = ((unsigned long long)hi << 32) | lo;
        best[mi][v] = o < best[mi][v] ? o : best[mi][v];
      }
  if (r == 0) {                                      // lane 0 -> rows 0..7, lane 16 -> rows 8..15
#pragma unroll
    for (int mi = 0; mi < 2; ++mi)
#pragma unroll
      for (int v = 0; v < 8; ++v) {
        int grow = (itm * 2 + mi) * 16 + v + 8 * h;
        partial[(size_t)grow * PSLOTS + chunk * 4 + w] = best[mi][v];
      }
  }
}

// ---------- K6: final argmin over partials + gather un-normalized anchors ----------
__global__ void k_gather(const unsigned long long* __restrict__ partial,
                         const float* __restrict__ anchors, float* __restrict__ out) {
  __shared__ int sidx;
  int row = blockIdx.x, tid = threadIdx.x;
  if (tid == 0) {
    unsigned long long best = 0xFFFFFFFFFFFFFFFFull;
    for (int p = 0; p < PSLOTS; ++p) {
      unsigned long long u = partial[(size_t)row * PSLOTS + p];
      best = u < best ? u : best;
    }
    sidx = (int)(unsigned)(best & 0xFFFFFFFFull);
  }
  __syncthreads();
  int idx = sidx;
  for (int c = tid; c < Z; c += blockDim.x)
    out[(size_t)row * Z + c] = anchors[(size_t)idx * Z + c];
}

extern "C" void kernel_launch(void* const* d_in, const int* in_sizes, int n_in,
                              void* d_out, int out_size, void* d_ws, size_t ws_size,
                              hipStream_t stream) {
  const float* idv = (const float*)d_in[0];   // [B, 512]
  const float* anc = (const float*)d_in[1];   // [NA, 512]
  float* out = (float*)d_out;                 // [B, 512]

  int B  = in_sizes[0] / Z;                   // 4096
  int NA = in_sizes[1] / Z;                   // 10000
  int BT = B / 16, AT = NA / 16;              // 256, 625

  // carve scratch (all offsets 256B aligned; ~68 MB total)
  char* p = (char*)d_ws;
  auto carve = [&](size_t bytes) { char* q = p; p += (bytes + 255) & ~(size_t)255; return q; };
  bf16_t* xb   = (bf16_t*)carve((size_t)B * Z * 2);
  bf16_t* xT   = (bf16_t*)carve((size_t)Z * B * 2);
  bf16_t* anb  = (bf16_t*)carve((size_t)NA * Z * 2);
  bf16_t* mask = (bf16_t*)carve((size_t)B * B * 2);
  float*  adjf = (float*)carve((size_t)B * Z * 4);
  bf16_t* adjb = (bf16_t*)carve((size_t)B * Z * 2);
  unsigned long long* partial =
      (unsigned long long*)carve((size_t)B * PSLOTS * sizeof(unsigned long long));
  (void)ws_size; (void)n_in; (void)out_size;

  k_rownorm_bf16<<<B, 256, 0, stream>>>(idv, xb, Z);
  k_rownorm_bf16<<<NA, 256, 0, stream>>>(anc, anb, Z);
  k_transpose<<<(B * Z + 255) / 256, 256, 0, stream>>>(xb, xT, B, Z);

  int sm_waves = (BT / 4) * (BT / 2);         // 8192 waves, 64x32 tile each
  k_self_mask<<<(sm_waves + 3) / 4, 128, 0, stream>>>(xb, mask, B, BT);

  int mx_waves = (BT / 4) * (ZT / 2);         // 1024 waves
  k_mask_x<<<(mx_waves + 3) / 4, 128, 0, stream>>>(mask, xT, adjf, B, BT);

  k_div_cvt<<<B, 256, 0, stream>>>(mask, adjf, adjb, B);

  int tpc = (AT + CH - 1) / CH;               // 25 anchor-tiles per chunk
  k_anchor_argmin<<<dim3(CH, BT / 2), 128, 0, stream>>>(adjb, anb, partial, AT, tpc);
  k_gather<<<B, 256, 0, stream>>>(partial, anc, out);
}
